// ManifoldAttention_41815801594183
// MI455X (gfx1250) — matrix operand order repaired and checked by reading, not compiled, not run on hardware
//
#include <hip/hip_runtime.h>
#include <hip/hip_bf16.h>

// ---------------------------------------------------------------------------
// ManifoldAttention on MI455X (gfx1250, wave32, WMMA + async tensor-to-LDS)
//
// ~172 GFLOP vs ~130 MB HBM traffic when attention is fused => compute-bound
// => all GEMMs via v_wmma_f32_16x16x32_bf16 (bf16 operands, fp32 accum).
// Flash-attention tiling keeps the 4096x4096 score matrix out of HBM.
// K/V tiles stream into LDS with GLOBAL_LOAD_ASYNC_TO_LDS_B128, double-
// buffered and overlapped with WMMA compute (ASYNCcnt threshold waits).
// ---------------------------------------------------------------------------

typedef __attribute__((ext_vector_type(16))) __bf16 v16bf;
typedef __attribute__((ext_vector_type(8)))  __bf16 v8bf;
typedef __attribute__((ext_vector_type(8)))  float  v8f;

#define BATCH   4
#define SEQ     4096
#define DMODEL  512
#define NTOK    (BATCH * SEQ)          // 16384
#define LN_EPS  1e-5f
#define NEG_INF (-1e30f)
#define SM_SCALE 0.044194173824159216f // 1/sqrt(512)

// --- WMMA fragment helpers --------------------------------------------------
// A-fragment (16x32 bf16): lane L<16 row L, elems 0..7 = K(c..c+7),
// 8..15 = K(c+16..c+23); lanes >=16 shifted +8 in K. rowptr pre-offset by
// (c + 8*(lane>>4)).
__device__ __forceinline__ v16bf load_a_frag(const __bf16* rowptr) {
    v8bf lo = *(const v8bf*)(rowptr);
    v8bf hi = *(const v8bf*)(rowptr + 16);
    v16bf a;
#pragma unroll
    for (int i = 0; i < 8; ++i) { a[i] = lo[i]; a[i + 8] = hi[i]; }
    return a;
}

// B-fragment (32x16 bf16): lane L<16 column N=L, elems 0..7 = K(c..c+7),
// 8..15 = K(c+16..c+23); lanes >=16 shifted +8 in K. colptr pre-offset by
// (c + 8*(lane>>4)).
__device__ __forceinline__ v16bf load_b_frag(const __bf16* colptr) {
    v8bf lo = *(const v8bf*)(colptr);
    v8bf hi = *(const v8bf*)(colptr + 16);
    v16bf b;
#pragma unroll
    for (int i = 0; i < 8; ++i) { b[i] = lo[i]; b[i + 8] = hi[i]; }
    return b;
}

__device__ __forceinline__ v8f v8f_zero() {
    v8f z;
#pragma unroll
    for (int i = 0; i < 8; ++i) z[i] = 0.f;
    return z;
}

// --- CDNA5 async global->LDS copy (ASYNCcnt-tracked, no VGPR staging) ------
// Generic pointers to LDS carry the LDS offset in their low 32 bits.
__device__ __forceinline__ void async_copy_b128(void* lds_dst, const void* gsrc) {
    unsigned loff = (unsigned)(uintptr_t)lds_dst;
    asm volatile("global_load_async_to_lds_b128 %0, %1, off"
                 :: "v"(loff), "v"(gsrc)
                 : "memory");
}
#define WAIT_ASYNC_LE32() asm volatile("s_wait_asynccnt 32" ::: "memory")
#define WAIT_ASYNC_0()    asm volatile("s_wait_asynccnt 0"  ::: "memory")

// --- Kernel 1: fp32 weight -> transposed bf16 ------------------------------
__global__ void transpose_bf16_kernel(const float* __restrict__ W,
                                      __bf16* __restrict__ WT,
                                      int R, int Ccols) {
    int idx = blockIdx.x * blockDim.x + threadIdx.x;
    if (idx < R * Ccols) {
        int r = idx / Ccols, n = idx % Ccols;
        WT[(size_t)n * R + r] = (__bf16)W[idx];
    }
}

// --- Kernel 2: LayerNorm + QKV projection (WMMA GEMM) ----------------------
// 256 threads = 8 waves per 16 token rows. V is written TRANSPOSED
// ([B][d][T]) so the attention kernel can async-stream V tiles with no
// transpose on the consume side.
__global__ void ln_qkv_kernel(const float* __restrict__ x,
                              const float* __restrict__ gamma,
                              const float* __restrict__ beta,
                              const __bf16* __restrict__ WqkvT, // [1536][512]
                              __bf16* __restrict__ qb,          // [B*T][512]
                              __bf16* __restrict__ kb,          // [B*T][512]
                              __bf16* __restrict__ vbT) {       // [B][512][T]
    __shared__ __bf16 Xs[16 * 520];
    const int m0  = blockIdx.x * 16;
    const int tid = threadIdx.x;

    // LayerNorm: 16 lanes per row, width-16 shuffle butterflies (wave32-safe).
    {
        const int r  = tid >> 4;
        const int c0 = tid & 15;
        const float* xr = x + (size_t)(m0 + r) * DMODEL;
        float vals[32];
        float s = 0.f, ss = 0.f;
#pragma unroll
        for (int k = 0; k < 32; ++k) {
            float v = xr[c0 + (k << 4)];
            vals[k] = v; s += v; ss += v * v;
        }
#pragma unroll
        for (int off = 8; off >= 1; off >>= 1) {
            s  += __shfl_xor(s,  off, 16);
            ss += __shfl_xor(ss, off, 16);
        }
        const float mu   = s * (1.f / DMODEL);
        const float var  = ss * (1.f / DMODEL) - mu * mu;
        const float rstd = rsqrtf(var + LN_EPS);
#pragma unroll
        for (int k = 0; k < 32; ++k) {
            const int c = c0 + (k << 4);
            float xn = (vals[k] - mu) * rstd * gamma[c] + beta[c];
            Xs[r * 520 + c] = (__bf16)xn;
        }
    }
    __syncthreads();

    // GEMM 16x1536, K=512. Wave w covers output cols [w*192, w*192+192).
    const int wave = tid >> 5, lane = tid & 31;
    const int al_row = lane & 15, al_off = (lane >> 4) << 3;
    const int bn = lane & 15,     bk     = (lane >> 4) << 3;
    const int row_off = (lane >> 4) << 3;

    v8f acc[12];
#pragma unroll
    for (int t = 0; t < 12; ++t) acc[t] = v8f_zero();

    for (int kk = 0; kk < 16; ++kk) {
        v16bf a = load_a_frag(&Xs[al_row * 520 + kk * 32 + al_off]);
        v16bf bfr[12];
#pragma unroll
        for (int t = 0; t < 12; ++t) {
            const int ncol = wave * 192 + t * 16 + bn;
            bfr[t] = load_b_frag(WqkvT + (size_t)ncol * DMODEL + kk * 32 + bk);
        }
#pragma unroll
        for (int t = 0; t < 12; ++t)
            acc[t] = __builtin_amdgcn_wmma_f32_16x16x32_bf16(
                false, a, false, bfr[t], (short)0, acc[t], false, false);
    }

    const int bbatch = m0 >> 12;        // token tile is within one batch
    const int s_base = (m0 & (SEQ - 1));
#pragma unroll
    for (int t = 0; t < 12; ++t) {
        const int col = wave * 192 + t * 16 + bn;
#pragma unroll
        for (int e = 0; e < 8; ++e) {
            const int row = m0 + e + row_off;
            __bf16 v = (__bf16)acc[t][e];
            if (col < DMODEL) {
                qb[(size_t)row * DMODEL + col] = v;
            } else if (col < 2 * DMODEL) {
                kb[(size_t)row * DMODEL + (col - DMODEL)] = v;
            } else {
                const int c = col - 2 * DMODEL;
                const int s = s_base + e + row_off;
                vbT[((size_t)bbatch * DMODEL + c) * SEQ + s] = v;
            }
        }
    }
}

// --- Kernel 3: causal flash attention --------------------------------------
// 128 threads = 4 waves own 64 query rows (16 per wave). Key tiles Bc=32,
// double-buffered in LDS via async copies (exactly 32 async b128 per thread
// per tile => s_wait_asynccnt 32 gates the pipeline).
// LDS: Q 64x520 | 2x K 32x520 | 2x V^T 512x40 | P scratch 4x16x40 (bf16)
__global__ void flash_attn_kernel(const __bf16* __restrict__ qb,
                                  const __bf16* __restrict__ kb,
                                  const __bf16* __restrict__ vbT,
                                  __bf16* __restrict__ ab) {
    extern __shared__ char smem[];
    __bf16* Qs = (__bf16*)smem;            // 64*520
    __bf16* Ks = Qs + 64 * 520;            // 2 * 32*520
    __bf16* Vt = Ks + 2 * 32 * 520;        // 2 * 512*40
    __bf16* Ps = Vt + 2 * 512 * 40;        // 4 * 16*40

    const int b   = blockIdx.y;
    const int m0  = blockIdx.x * 64;
    const int tid = threadIdx.x;           // 128
    const int wave = tid >> 5, lane = tid & 31;
    const int al_row = lane & 15, al_off = (lane >> 4) << 3;
    const int bn = lane & 15,     bk     = (lane >> 4) << 3;
    const int row_off = (lane >> 4) << 3;
    const size_t base = (size_t)b * SEQ * DMODEL; // works for qb/kb and vbT

    // Stage Q (64x512) asynchronously: 32 b128 per thread.
    for (int ch = tid; ch < 4096; ch += 128) {
        const int r = ch >> 6, c8 = (ch & 63) << 3;
        async_copy_b128(&Qs[r * 520 + c8],
                        qb + base + (size_t)(m0 + r) * DMODEL + c8);
    }

    const int nj = blockIdx.x * 2 + 2;     // causal: key tiles up to diagonal

    // Issue tile 0 (K: 16 ops, V: 16 ops per thread).
    {
        __bf16* KsB = Ks; __bf16* VtB = Vt;
        for (int ch = tid; ch < 2048; ch += 128) {
            const int r = ch >> 6, c8 = (ch & 63) << 3;
            async_copy_b128(&KsB[r * 520 + c8],
                            kb + base + (size_t)r * DMODEL + c8);
        }
        for (int ch = tid; ch < 2048; ch += 128) {
            const int c = ch >> 2, s8 = (ch & 3) << 3;
            async_copy_b128(&VtB[c * 40 + s8],
                            vbT + base + (size_t)c * SEQ + s8);
        }
    }

    v8f o[32];
#pragma unroll
    for (int t = 0; t < 32; ++t) o[t] = v8f_zero();
    float mst[8], lst[8];
#pragma unroll
    for (int e = 0; e < 8; ++e) { mst[e] = NEG_INF; lst[e] = 0.f; }

    for (int j = 0; j < nj; ++j) {
        const int j0 = j * 32;
        const int buf = j & 1;
        __bf16* KsB = Ks + buf * (32 * 520);
        __bf16* VtB = Vt + buf * (512 * 40);

        __syncthreads(); // all waves done reading the buffer we now refill

        if (j + 1 < nj) { // prefetch next tile into the other buffer
            __bf16* KsN = Ks + (buf ^ 1) * (32 * 520);
            __bf16* VtN = Vt + (buf ^ 1) * (512 * 40);
            const int j1 = j0 + 32;
            for (int ch = tid; ch < 2048; ch += 128) {
                const int r = ch >> 6, c8 = (ch & 63) << 3;
                async_copy_b128(&KsN[r * 520 + c8],
                                kb + base + (size_t)(j1 + r) * DMODEL + c8);
            }
            for (int ch = tid; ch < 2048; ch += 128) {
                const int c = ch >> 2, s8 = (ch & 3) << 3;
                async_copy_b128(&VtN[c * 40 + s8],
                                vbT + base + (size_t)c * SEQ + j1 + s8);
            }
            WAIT_ASYNC_LE32(); // in-order: tile j (and Q on j==0) has landed
        } else {
            WAIT_ASYNC_0();
        }
        __syncthreads();

        // S = Q K^T : 16x32 per wave (2 N-tiles, 16 k-steps of 32).
        v8f sacc[2];
        sacc[0] = v8f_zero(); sacc[1] = v8f_zero();
        for (int kk = 0; kk < 16; ++kk) {
            v16bf a  = load_a_frag(&Qs[(wave * 16 + al_row) * 520 + kk * 32 + al_off]);
            v16bf b0 = load_b_frag(&KsB[(bn)      * 520 + kk * 32 + bk]);
            v16bf b1 = load_b_frag(&KsB[(16 + bn) * 520 + kk * 32 + bk]);
            sacc[0] = __builtin_amdgcn_wmma_f32_16x16x32_bf16(
                false, a, false, b0, (short)0, sacc[0], false, false);
            sacc[1] = __builtin_amdgcn_wmma_f32_16x16x32_bf16(
                false, a, false, b1, (short)0, sacc[1], false, false);
        }

        // Online softmax. C/D layout: element e -> row e+row_off; lanes = cols.
        float mnew[8];
#pragma unroll
        for (int e = 0; e < 8; ++e) mnew[e] = mst[e];
#pragma unroll
        for (int t = 0; t < 2; ++t)
#pragma unroll
            for (int e = 0; e < 8; ++e) {
                const int col = j0 + t * 16 + bn;
                const int row = m0 + wave * 16 + e + row_off;
                float s = sacc[t][e] * SM_SCALE;
                s = (col <= row) ? s : NEG_INF;
                sacc[t][e] = s;
                mnew[e] = fmaxf(mnew[e], s);
            }
#pragma unroll
        for (int off = 8; off >= 1; off >>= 1)
#pragma unroll
            for (int e = 0; e < 8; ++e)
                mnew[e] = fmaxf(mnew[e], __shfl_xor(mnew[e], off, 16));

        float alpha[8], rsum[8];
#pragma unroll
        for (int e = 0; e < 8; ++e) {
            alpha[e] = __expf(mst[e] - mnew[e]);
            mst[e] = mnew[e];
            rsum[e] = 0.f;
        }
#pragma unroll
        for (int t = 0; t < 2; ++t)
#pragma unroll
            for (int e = 0; e < 8; ++e) {
                float p = __expf(sacc[t][e] - mnew[e]);
                sacc[t][e] = p;
                rsum[e] += p;
            }
#pragma unroll
        for (int off = 8; off >= 1; off >>= 1)
#pragma unroll
            for (int e = 0; e < 8; ++e)
                rsum[e] += __shfl_xor(rsum[e], off, 16);
#pragma unroll
        for (int e = 0; e < 8; ++e) lst[e] = lst[e] * alpha[e] + rsum[e];
#pragma unroll
        for (int t = 0; t < 32; ++t)
#pragma unroll
            for (int e = 0; e < 8; ++e) o[t][e] *= alpha[e];

        // Re-layout P (C/D layout) -> A-fragment layout via per-wave LDS.
#pragma unroll
        for (int t = 0; t < 2; ++t)
#pragma unroll
            for (int e = 0; e < 8; ++e)
                Ps[wave * 640 + (e + row_off) * 40 + t * 16 + bn] =
                    (__bf16)sacc[t][e];
        asm volatile("s_wait_dscnt 0" ::: "memory"); // same-wave LDS hand-off

        // O += P * V : one k-step of 32 keys, 32 N-tiles over d=512,
        // B-fragments preloaded in groups of 8 to batch the waits.
        {
            v16bf a = load_a_frag(&Ps[wave * 640 + al_row * 40 + al_off]);
#pragma unroll
            for (int g = 0; g < 4; ++g) {
                v16bf bf[8];
#pragma unroll
                for (int i = 0; i < 8; ++i) {
                    const int nt = g * 8 + i;
                    bf[i] = load_b_frag(&VtB[(nt * 16 + bn) * 40 + bk]);
                }
#pragma unroll
                for (int i = 0; i < 8; ++i)
                    o[g * 8 + i] = __builtin_amdgcn_wmma_f32_16x16x32_bf16(
                        false, a, false, bf[i], (short)0, o[g * 8 + i],
                        false, false);
            }
        }
    }

    // Normalize and emit bf16 attention output.
    float rl[8];
#pragma unroll
    for (int e = 0; e < 8; ++e) rl[e] = 1.f / lst[e];
#pragma unroll
    for (int nt = 0; nt < 32; ++nt)
#pragma unroll
        for (int e = 0; e < 8; ++e) {
            const int row = m0 + wave * 16 + e + row_off;
            const int col = nt * 16 + bn;
            ab[base + (size_t)row * DMODEL + col] = (__bf16)(o[nt][e] * rl[e]);
        }
}

// --- Kernel 4: output projection (WMMA GEMM, fp32 out) ---------------------
__global__ void proj_kernel(const __bf16* __restrict__ ab,
                            const __bf16* __restrict__ WprojT, // [512][512]^T
                            float* __restrict__ out) {
    __shared__ __bf16 As[16 * 520];
    const int m0  = blockIdx.x * 16;
    const int tid = threadIdx.x; // 256
    for (int ch = tid; ch < 16 * 64; ch += 256) {
        const int r = ch >> 6, c8 = (ch & 63) << 3;
        *(v8bf*)&As[r * 520 + c8] =
            *(const v8bf*)&ab[(size_t)(m0 + r) * DMODEL + c8];
    }
    __syncthreads();

    const int wave = tid >> 5, lane = tid & 31;
    const int al_row = lane & 15, al_off = (lane >> 4) << 3;
    const int bn = lane & 15,     bk     = (lane >> 4) << 3;
    const int row_off = (lane >> 4) << 3;

    v8f acc[4];
#pragma unroll
    for (int t = 0; t < 4; ++t) acc[t] = v8f_zero();

    for (int kk = 0; kk < 16; ++kk) {
        v16bf a = load_a_frag(&As[al_row * 520 + kk * 32 + al_off]);
        v16bf bfr[4];
#pragma unroll
        for (int t = 0; t < 4; ++t) {
            const int ncol = wave * 64 + t * 16 + bn;
            bfr[t] = load_b_frag(WprojT + (size_t)ncol * DMODEL + kk * 32 + bk);
        }
#pragma unroll
        for (int t = 0; t < 4; ++t)
            acc[t] = __builtin_amdgcn_wmma_f32_16x16x32_bf16(
                false, a, false, bfr[t], (short)0, acc[t], false, false);
    }
#pragma unroll
    for (int t = 0; t < 4; ++t) {
        const int col = wave * 64 + t * 16 + bn;
#pragma unroll
        for (int e = 0; e < 8; ++e)
            out[(size_t)(m0 + e + row_off) * DMODEL + col] = acc[t][e];
    }
}

// ---------------------------------------------------------------------------
extern "C" void kernel_launch(void* const* d_in, const int* in_sizes, int n_in,
                              void* d_out, int out_size, void* d_ws, size_t ws_size,
                              hipStream_t stream) {
    const float* x     = (const float*)d_in[0];
    // d_in[1] = causal mask: recomputed analytically, unused.
    const float* gamma = (const float*)d_in[2];
    const float* beta  = (const float*)d_in[3];
    const float* Wqkv  = (const float*)d_in[4];
    const float* Wproj = (const float*)d_in[5];
    float* out = (float*)d_out;

    // Workspace (bf16): Q, K, V^T, attn_out (16MB each) + transposed weights.
    const size_t MAT = (size_t)NTOK * DMODEL;
    __bf16* qb     = (__bf16*)d_ws;
    __bf16* kb     = qb + MAT;
    __bf16* vbT    = kb + MAT;          // [B][512][T]
    __bf16* ab     = vbT + MAT;
    __bf16* WqkvT  = ab + MAT;
    __bf16* WprojT = WqkvT + (size_t)DMODEL * 3 * DMODEL;

    transpose_bf16_kernel<<<(DMODEL * 3 * DMODEL + 255) / 256, 256, 0, stream>>>(
        Wqkv, WqkvT, DMODEL, 3 * DMODEL);
    transpose_bf16_kernel<<<(DMODEL * DMODEL + 255) / 256, 256, 0, stream>>>(
        Wproj, WprojT, DMODEL, DMODEL);

    ln_qkv_kernel<<<NTOK / 16, 256, 0, stream>>>(x, gamma, beta, WqkvT,
                                                 qb, kb, vbT);

    const size_t lds_bytes =
        (size_t)(64 * 520 + 2 * 32 * 520 + 2 * 512 * 40 + 4 * 16 * 40)
        * sizeof(__bf16); // 220,160 B of the 320 KB WGP pool
    flash_attn_kernel<<<dim3(SEQ / 64, BATCH), 128, lds_bytes, stream>>>(
        qb, kb, vbT, ab);

    proj_kernel<<<NTOK / 16, 256, 0, stream>>>(ab, WprojT, out);
}
